// MultiHeadedAttention_84499186581895
// MI455X (gfx1250) — matrix operand order repaired and checked by reading, not compiled, not run on hardware
//
#include <hip/hip_runtime.h>
#include <hip/hip_bf16.h>

#define S_LEN 2048
#define NHEADS 16
#define HS 64
#define BATCH 2
#define EMB 1024

typedef __bf16 bf16_t;
typedef __attribute__((ext_vector_type(16))) __bf16 v16bf;
typedef __attribute__((ext_vector_type(8)))  __bf16 v8bf;
typedef __attribute__((ext_vector_type(8)))  float  v8f;
typedef __attribute__((ext_vector_type(4)))  float  v4f;
typedef __attribute__((ext_vector_type(4)))  int    v4i;

union V16U { v16bf v; v8bf h[2]; };

#if defined(__gfx1250__) && __has_builtin(__builtin_amdgcn_global_load_async_to_lds_b128)
#define HAVE_ASYNC_LDS 1
typedef __attribute__((address_space(1))) v4i g_v4i;
typedef __attribute__((address_space(3))) v4i l_v4i;
#endif

// Copy 32 bytes global -> LDS (async-to-LDS when available, sync fallback otherwise).
__device__ __forceinline__ void cp32_g2l(const bf16_t* g, bf16_t* l) {
#ifdef HAVE_ASYNC_LDS
  __builtin_amdgcn_global_load_async_to_lds_b128((g_v4i*)g, (l_v4i*)l, 0, 0);
  __builtin_amdgcn_global_load_async_to_lds_b128((g_v4i*)g, (l_v4i*)l, 16, 0);
#else
  *(v8bf*)l       = *(const v8bf*)g;
  *(v8bf*)(l + 8) = *(const v8bf*)(g + 8);
#endif
}

// Wait until at most N async global->LDS ops remain outstanding (in-order retire).
template <int N>
__device__ __forceinline__ void wait_g2l() {
#ifdef HAVE_ASYNC_LDS
#if __has_builtin(__builtin_amdgcn_s_wait_asynccnt)
  __builtin_amdgcn_s_wait_asynccnt(N);
#else
  asm volatile("s_wait_asynccnt %0" ::"i"(N) : "memory");
#endif
#endif
}

__device__ __forceinline__ v8f wmma_bf16(v16bf a, v16bf b, v8f c) {
  // D = A(16x32 bf16) * B(32x16 bf16) + C(16x16 f32)
  return __builtin_amdgcn_wmma_f32_16x16x32_bf16(false, a, false, b, (short)0, c,
                                                 false, false);
}

// A-fragment (16x32 bf16, row-major source):
// lane L: row = L&15; K base = (L<16)?0:8; elems 0..7 -> K=kb..kb+7, 8..15 -> K=16+kb..
__device__ __forceinline__ v16bf load_a_frag(const bf16_t* base, int stride, int lane) {
  const int row = lane & 15;
  const int kb  = (lane < 16) ? 0 : 8;
  const bf16_t* p = base + (size_t)row * stride + kb;
  V16U u;
  u.h[0] = *(const v8bf*)(p);
  u.h[1] = *(const v8bf*)(p + 16);
  return u.v;
}

// B-fragment (32x16 bf16) from a transposed [n][k] source:
// lane L: col n = L&15; K base = (L<16)?0:8; elems 0..7 -> K=kb..kb+7, 8..15 -> K=16+kb..
__device__ __forceinline__ v16bf load_b_frag(const bf16_t* baseT, int strideT, int lane) {
  const int n  = lane & 15;
  const int kb = (lane < 16) ? 0 : 8;
  const bf16_t* p = baseT + (size_t)n * strideT + kb;
  V16U u;
  u.h[0] = *(const v8bf*)(p);
  u.h[1] = *(const v8bf*)(p + 16);
  return u.v;
}

// ---------------------------------------------------------------------------
// Prep: WT[b][c][r] = (bf16)W[b][r][c]
__global__ void __launch_bounds__(256) wtrans_kernel(const float* __restrict__ W,
                                                     bf16_t* __restrict__ WT,
                                                     int rows, int cols, int batch) {
  const size_t idx = (size_t)blockIdx.x * 256 + threadIdx.x;
  const size_t total = (size_t)batch * rows * cols;
  if (idx >= total) return;
  const int c  = (int)(idx % cols);
  const int r  = (int)((idx / cols) % rows);
  const int bb = (int)(idx / ((size_t)cols * rows));
  WT[((size_t)bb * cols + c) * rows + r] = (bf16_t)W[idx];
}

// ---------------------------------------------------------------------------
// Per-head projection: out = X[b] (S x 64) * W[h] (64 x 64), bf16 result.
// transposed_store==0 : out layout [B,H,S,64] (for Q and K)
// transposed_store==1 : out layout [B,H,64,S] (for V, feeds P*V B-fragments)
__global__ void __launch_bounds__(128) proj_kernel(const float* __restrict__ X,
                                                   const bf16_t* __restrict__ WT,
                                                   bf16_t* __restrict__ out,
                                                   int transposed_store) {
  const int lane = threadIdx.x & 31;
  const int wave = threadIdx.x >> 5;
  const int tile = blockIdx.x * 4 + wave;
  const int ST = S_LEN / 16;
  const int st = tile % ST;
  const int h  = (tile / ST) % NHEADS;
  const int b  = tile / (ST * NHEADS);
  const int s0 = st * 16;

  // Build A-fragments from fp32 input, converting to bf16 in-register.
  const int arow = lane & 15;
  const int akb  = (lane < 16) ? 0 : 8;
  const float* xp = X + ((size_t)b * S_LEN + s0 + arow) * HS;
  v16bf a[2];
#pragma unroll
  for (int f = 0; f < 2; ++f) {
    const float* p0 = xp + f * 32 + akb;
    const float* p1 = xp + f * 32 + 16 + akb;
#pragma unroll
    for (int e = 0; e < 8; ++e) {
      a[f][e]     = (bf16_t)p0[e];
      a[f][e + 8] = (bf16_t)p1[e];
    }
  }

  const bf16_t* wt = WT + (size_t)h * HS * HS;
  v8f c[4] = {};
#pragma unroll
  for (int nt = 0; nt < 4; ++nt)
#pragma unroll
    for (int f = 0; f < 2; ++f) {
      v16bf bb = load_b_frag(wt + (size_t)(nt * 16) * HS + f * 32, HS, lane);
      c[nt] = wmma_bf16(a[f], bb, c[nt]);
    }

  const int row_off = (lane < 16) ? 0 : 8;
  const int colb = lane & 15;

  if (transposed_store) {
    // C-frag lane holds 8 consecutive rows of one column -> contiguous in [col][s].
#pragma unroll
    for (int nt = 0; nt < 4; ++nt) {
      v8bf t;
#pragma unroll
      for (int r = 0; r < 8; ++r) t[r] = (bf16_t)c[nt][r];
      const int col = nt * 16 + colb;
      bf16_t* dst = out + (((size_t)b * NHEADS + h) * HS + col) * S_LEN + s0 + row_off;
      *(v8bf*)dst = t;
    }
  } else {
    __shared__ bf16_t tileb[4][16][64];
#pragma unroll
    for (int nt = 0; nt < 4; ++nt)
#pragma unroll
      for (int r = 0; r < 8; ++r)
        tileb[wave][row_off + r][nt * 16 + colb] = (bf16_t)c[nt][r];
    asm volatile("s_wait_dscnt 0" ::: "memory");
#pragma unroll
    for (int pass = 0; pass < 4; ++pass) {
      const int rr  = pass * 4 + (lane >> 3);
      const int seg = lane & 7;
      v8bf t = *(const v8bf*)&tileb[wave][rr][seg * 8];
      bf16_t* dst = out + (((size_t)b * NHEADS + h) * S_LEN + s0 + rr) * HS + seg * 8;
      *(v8bf*)dst = t;
    }
  }
}

// ---------------------------------------------------------------------------
// Flash attention: 8 waves/block share one (b,h); each wave owns 16 query rows.
// K/V tiles (64-wide) are double-buffered in LDS via async-to-LDS DMA so the
// next tile's copies overlap the current tile's WMMA + softmax work.
__global__ void __launch_bounds__(256) flash_kernel(const bf16_t* __restrict__ qg,
                                                    const bf16_t* __restrict__ kg,
                                                    const bf16_t* __restrict__ vtg,
                                                    bf16_t* __restrict__ og) {
  const int tid  = threadIdx.x;
  const int lane = tid & 31;
  const int wave = tid >> 5;
  const int QT = S_LEN / 128;
  const int qt = blockIdx.x % QT;
  const int h  = (blockIdx.x / QT) % NHEADS;
  const int b  = blockIdx.x / (QT * NHEADS);
  const int q0 = qt * 128 + wave * 16;

  const bf16_t* qb = qg  + ((size_t)b * NHEADS + h) * S_LEN * HS;
  const bf16_t* kk = kg  + ((size_t)b * NHEADS + h) * S_LEN * HS;
  const bf16_t* vt = vtg + ((size_t)b * NHEADS + h) * HS * S_LEN;

  v16bf aq[2];
#pragma unroll
  for (int f = 0; f < 2; ++f)
    aq[f] = load_a_frag(qb + (size_t)q0 * HS + f * 32, HS, lane);

  const int row_off = (lane < 16) ? 0 : 8;
  const int colb = lane & 15;
  const float RS = 0.022097086912079612f;  // 1/sqrt(2048); ref divides (s+mask)/sqrt(Sk)
  const float NEG = -3.0e38f;

  float mrow[8], lrow[8];
  v8f acc[4] = {};
#pragma unroll
  for (int r = 0; r < 8; ++r) { mrow[r] = NEG; lrow[r] = 0.f; }

  __shared__ bf16_t kbuf[2][64 * 64];    // K tiles  [k-row][hd], row-major, 2 x 8KB
  __shared__ bf16_t vbuf[2][64 * 64];    // V^T tiles [hd-col][k], 2 x 8KB
  __shared__ bf16_t plds[8][16][64];     // per-wave P tiles; reused for output staging

  const int my_tiles  = (q0 + 79) / 64;  // 64-wide causal tiles this wave needs
  const int blk_tiles = qt * 2 + 2;      // tiles the whole block stages

  // Each thread stages 32B of K and 32B of V^T per tile (4 async ops / thread / tile).
  const int vcol = tid >> 2, vseg = tid & 3;

  // prologue: stage tile 0 into buffer 0
  cp32_g2l(kk + (size_t)0 * HS + tid * 16, &kbuf[0][tid * 16]);
  cp32_g2l(vt + (size_t)vcol * S_LEN + 0 + vseg * 16, &vbuf[0][vcol * 64 + vseg * 16]);

  for (int t = 0; t < blk_tiles; ++t) {
    const int k0  = t * 64;
    const int cur = t & 1;

    if (t + 1 < blk_tiles) {
      // issue next tile's DMA into the other buffer, then wait only for tile t
      const int k1 = k0 + 64;
      cp32_g2l(kk + (size_t)k1 * HS + tid * 16, &kbuf[cur ^ 1][tid * 16]);
      cp32_g2l(vt + (size_t)vcol * S_LEN + k1 + vseg * 16,
               &vbuf[cur ^ 1][vcol * 64 + vseg * 16]);
      wait_g2l<4>();   // 4 younger ops (tile t+1) may remain in flight
    } else {
      wait_g2l<0>();
    }
    __syncthreads();   // every wave's tile-t data is now in LDS

    if (t < my_tiles) {
      // S = Q * K^T over a 16x64 score tile (4 column fragments)
      v8f sv[4] = {};
#pragma unroll
      for (int f = 0; f < 2; ++f)
#pragma unroll
        for (int nt = 0; nt < 4; ++nt) {
          v16bf bk = load_b_frag(&kbuf[cur][(nt * 16) * 64 + f * 32], 64, lane);
          sv[nt] = wmma_bf16(aq[f], bk, sv[nt]);
        }

      // scale + causal mask (skip per-element compares for interior tiles)
      const bool need_mask = (k0 + 63) > q0;
      float tmax[8], psum[8], alpha[8];
#pragma unroll
      for (int r = 0; r < 8; ++r) tmax[r] = NEG;
#pragma unroll
      for (int nt = 0; nt < 4; ++nt)
#pragma unroll
        for (int r = 0; r < 8; ++r) {
          float v = sv[nt][r] * RS;
          if (need_mask) {
            const int grow = q0 + row_off + r;
            if (k0 + nt * 16 + colb > grow) v = NEG;
          }
          sv[nt][r] = v;
          tmax[r] = fmaxf(tmax[r], v);
        }
      // row max across the 16 lanes holding each row
#pragma unroll
      for (int r = 0; r < 8; ++r)
#pragma unroll
        for (int xm = 1; xm < 16; xm <<= 1)
          tmax[r] = fmaxf(tmax[r], __shfl_xor(tmax[r], xm, 32));

      // online softmax update
#pragma unroll
      for (int r = 0; r < 8; ++r) {
        const float mn = fmaxf(mrow[r], tmax[r]);
        alpha[r] = __expf(mrow[r] - mn);
        mrow[r] = mn;
        psum[r] = 0.f;
      }
#pragma unroll
      for (int nt = 0; nt < 4; ++nt)
#pragma unroll
        for (int r = 0; r < 8; ++r) {
          const float p = __expf(sv[nt][r] - mrow[r]);
          sv[nt][r] = p;
          psum[r] += p;
        }
#pragma unroll
      for (int r = 0; r < 8; ++r) {
#pragma unroll
        for (int xm = 1; xm < 16; xm <<= 1)
          psum[r] += __shfl_xor(psum[r], xm, 32);
        lrow[r] = lrow[r] * alpha[r] + psum[r];
      }
#pragma unroll
      for (int nt = 0; nt < 4; ++nt)
#pragma unroll
        for (int r = 0; r < 8; ++r)
          acc[nt][r] *= alpha[r];

      // reshape P: C-fragment layout -> A-fragment layout via per-wave LDS tile
#pragma unroll
      for (int nt = 0; nt < 4; ++nt)
#pragma unroll
        for (int r = 0; r < 8; ++r)
          plds[wave][row_off + r][nt * 16 + colb] = (bf16_t)sv[nt][r];
      asm volatile("s_wait_dscnt 0" ::: "memory");
      v16bf ap0 = load_a_frag(&plds[wave][0][0], 64, lane);       // K 0..31
      v16bf ap1 = load_a_frag(&plds[wave][0][0] + 32, 64, lane);  // K 32..63

      // O += P * V  (V^T tile in LDS -> contiguous B fragments)
#pragma unroll
      for (int nt = 0; nt < 4; ++nt) {
        v16bf bv0 = load_b_frag(&vbuf[cur][(nt * 16) * 64 + 0], 64, lane);
        acc[nt] = wmma_bf16(ap0, bv0, acc[nt]);
        v16bf bv1 = load_b_frag(&vbuf[cur][(nt * 16) * 64 + 32], 64, lane);
        acc[nt] = wmma_bf16(ap1, bv1, acc[nt]);
      }
    }
    __syncthreads();   // buffer `cur` free before it is re-staged at t+2
  }

  float rl[8];
#pragma unroll
  for (int r = 0; r < 8; ++r) rl[r] = 1.0f / lrow[r];

  // normalize, bf16-ify, coalesced store into concat layout [B,S,EMB]
  // (reuse per-wave plds tile as staging; its last P use was in the final tile)
#pragma unroll
  for (int nt = 0; nt < 4; ++nt)
#pragma unroll
    for (int r = 0; r < 8; ++r)
      plds[wave][row_off + r][nt * 16 + colb] = (bf16_t)(acc[nt][r] * rl[r]);
  asm volatile("s_wait_dscnt 0" ::: "memory");
#pragma unroll
  for (int pass = 0; pass < 4; ++pass) {
    const int rr  = pass * 4 + (lane >> 3);
    const int seg = lane & 7;
    v8bf tv = *(const v8bf*)&plds[wave][rr][seg * 8];
    bf16_t* dst = og + ((size_t)b * S_LEN + q0 + rr) * EMB + h * HS + seg * 8;
    *(v8bf*)dst = tv;
  }
}

// ---------------------------------------------------------------------------
// Output projection: out[4096,1024] = concat(bf16) * Wo + bo, f32 result.
__global__ void __launch_bounds__(128) outproj_kernel(const bf16_t* __restrict__ og,
                                                      const bf16_t* __restrict__ WoT,
                                                      const float* __restrict__ bo,
                                                      float* __restrict__ out) {
  const int lane = threadIdx.x & 31;
  const int wave = threadIdx.x >> 5;
  const int tile = blockIdx.x * 4 + wave;
  const int NT = EMB / 64;  // 16
  const int ntile = tile % NT;
  const int mt = tile / NT;
  const int m0 = mt * 16;
  const int n0 = ntile * 64;

  v8f c[4] = {};
  for (int k0 = 0; k0 < EMB; k0 += 32) {
    v16bf a = load_a_frag(og + (size_t)m0 * EMB + k0, EMB, lane);
#pragma unroll
    for (int nt = 0; nt < 4; ++nt) {
      v16bf bb = load_b_frag(WoT + (size_t)(n0 + nt * 16) * EMB + k0, EMB, lane);
      c[nt] = wmma_bf16(a, bb, c[nt]);
    }
  }

  const int row_off = (lane < 16) ? 0 : 8;
  const int colb = lane & 15;
  __shared__ float ofl[4][16][64];
#pragma unroll
  for (int nt = 0; nt < 4; ++nt) {
    const float bias = bo[n0 + nt * 16 + colb];
#pragma unroll
    for (int r = 0; r < 8; ++r)
      ofl[wave][row_off + r][nt * 16 + colb] = c[nt][r] + bias;
  }
  asm volatile("s_wait_dscnt 0" ::: "memory");
#pragma unroll
  for (int pass = 0; pass < 8; ++pass) {
    const int rr  = pass * 2 + (lane >> 4);
    const int seg = lane & 15;
    v4f tv = *(const v4f*)&ofl[wave][rr][seg * 4];
    *(v4f*)(out + ((size_t)m0 + rr) * EMB + n0 + seg * 4) = tv;
  }
}

// ---------------------------------------------------------------------------
extern "C" void kernel_launch(void* const* d_in, const int* in_sizes, int n_in,
                              void* d_out, int out_size, void* d_ws, size_t ws_size,
                              hipStream_t stream) {
  (void)in_sizes; (void)n_in; (void)out_size; (void)ws_size;
  const float* x_k = (const float*)d_in[0];
  const float* x_v = (const float*)d_in[1];
  const float* x_q = (const float*)d_in[2];
  const float* Wk  = (const float*)d_in[3];
  const float* Wv  = (const float*)d_in[4];
  const float* Wq  = (const float*)d_in[5];
  const float* Wo  = (const float*)d_in[6];
  const float* bo  = (const float*)d_in[7];
  float* out = (float*)d_out;

  char* ws = (char*)d_ws;
  size_t off = 0;
  auto carve = [&](size_t bytes) -> char* {
    char* p = ws + off;
    off += (bytes + 255) & ~(size_t)255;
    return p;
  };
  bf16_t* WkT  = (bf16_t*)carve((size_t)NHEADS * HS * HS * sizeof(bf16_t));
  bf16_t* WvT  = (bf16_t*)carve((size_t)NHEADS * HS * HS * sizeof(bf16_t));
  bf16_t* WqT  = (bf16_t*)carve((size_t)NHEADS * HS * HS * sizeof(bf16_t));
  bf16_t* WoT  = (bf16_t*)carve((size_t)EMB * EMB * sizeof(bf16_t));
  bf16_t* qbf  = (bf16_t*)carve((size_t)BATCH * NHEADS * S_LEN * HS * sizeof(bf16_t));
  bf16_t* kbf  = (bf16_t*)carve((size_t)BATCH * NHEADS * S_LEN * HS * sizeof(bf16_t));
  bf16_t* vtbf = (bf16_t*)carve((size_t)BATCH * NHEADS * HS * S_LEN * sizeof(bf16_t));
  bf16_t* obf  = (bf16_t*)carve((size_t)BATCH * S_LEN * EMB * sizeof(bf16_t));

  // 1) weight transpose + bf16 downconvert
  wtrans_kernel<<<(NHEADS * HS * HS + 255) / 256, 256, 0, stream>>>(Wk, WkT, HS, HS, NHEADS);
  wtrans_kernel<<<(NHEADS * HS * HS + 255) / 256, 256, 0, stream>>>(Wv, WvT, HS, HS, NHEADS);
  wtrans_kernel<<<(NHEADS * HS * HS + 255) / 256, 256, 0, stream>>>(Wq, WqT, HS, HS, NHEADS);
  wtrans_kernel<<<(EMB * EMB + 255) / 256, 256, 0, stream>>>(Wo, WoT, EMB, EMB, 1);

  // 2) per-head Q/K/V projections (WMMA); V stored transposed
  const int proj_blocks = BATCH * NHEADS * (S_LEN / 16) / 4;  // 1024
  proj_kernel<<<proj_blocks, 128, 0, stream>>>(x_q, WqT, qbf, 0);
  proj_kernel<<<proj_blocks, 128, 0, stream>>>(x_k, WkT, kbf, 0);
  proj_kernel<<<proj_blocks, 128, 0, stream>>>(x_v, WvT, vtbf, 1);

  // 3) causal flash attention (WMMA QK^T + online softmax + WMMA PV,
  //    double-buffered async-to-LDS K/V staging)
  flash_kernel<<<BATCH * NHEADS * (S_LEN / 128), 256, 0, stream>>>(qbf, kbf, vtbf, obf);

  // 4) output projection + bias (WMMA), f32 result
  outproj_kernel<<<(BATCH * S_LEN / 16) * (EMB / 64) / 4, 128, 0, stream>>>(obf, WoT, bo, out);
}
